// NeuralODE_88399016886653
// MI455X (gfx1250) — matrix operand order repaired and checked by reading, not compiled, not run on hardware
//
#include <hip/hip_runtime.h>

typedef __attribute__((ext_vector_type(16))) __bf16 v16bf;
typedef __attribute__((ext_vector_type(8)))  __bf16 v8bf;
typedef __attribute__((ext_vector_type(8)))  float  v8f;
typedef __attribute__((ext_vector_type(4)))  float  v4f;

#define B_TOT 8192
#define T_LEN 65
#define D_DIM 32
#define H_DIM 128
#define TD    (T_LEN * D_DIM)   // 2080

#define SWAPX16 0x401f          // ds_swizzle: xor=0x10, or=0, and=0x1f -> lane ^= 16

__device__ __forceinline__ float fast_tanh(float x) {
#if __has_builtin(__builtin_amdgcn_tanhf)
    return __builtin_amdgcn_tanhf(x);
#else
    float e = __expf(2.0f * x);
    return 1.0f - 2.0f * __builtin_amdgcn_rcpf(e + 1.0f);
#endif
}

union Frag  { v16bf v; unsigned int u[8]; };
union Pack8 { v8bf  v; unsigned int u[4]; };

__device__ __forceinline__ unsigned int swz16(unsigned int v) {
    return (unsigned int)__builtin_amdgcn_ds_swizzle((int)v, SWAPX16);
}

// Build a 32(K) x 16(N) bf16 B operand from two C-layout f32 tiles.
// lo_t: values at k = 32k0 + 8*hh + r  (tile covering k in [32k0, 32k0+16))
// hi_t: values at k = 32k0 + 16 + 8*hh + r
// B-layout here: lane (n + 16*hh) holds k = 32k0 + 8*hh + 0..7 and 32k0 + 16 + 8*hh + 0..7.
// Own lane already holds both runs; no lane^16 exchange is needed (s0/s1 unused).
__device__ __forceinline__ v16bf build_B(v8f lo_t, v8f hi_t, bool hh) {
    Pack8 pa, pb;                               // <8 x f32> -> <8 x bf16>: v_cvt_pk_bf16_f32 x4
    pa.v = __builtin_convertvector(lo_t, v8bf);
    pb.v = __builtin_convertvector(hi_t, v8bf);
    Frag f;
#pragma unroll
    for (int j = 0; j < 4; ++j) {
        unsigned int s0 = swz16(pa.u[j]);
        unsigned int s1 = swz16(pb.u[j]);
        f.u[j]     = pa.u[j];             // elements 0..7  (k = 8*hh + 0..7)
        f.u[4 + j] = pb.u[j];             // elements 8..15 (k = 16 + 8*hh + 0..7)
    }
    return f.v;
}

__global__ __launch_bounds__(32, 1)
void node_dopri5_kernel(const float* __restrict__ x,
                        const float* __restrict__ tvec,
                        const float* __restrict__ W1,
                        const float* __restrict__ b1,
                        const float* __restrict__ W2,
                        const float* __restrict__ b2,
                        float* __restrict__ out)
{
    const int  lane = threadIdx.x;     // one wave32 per block
    const int  n    = lane & 15;       // batch row within tile (N index everywhere)
    const int  hh   = lane >> 4;       // lane-half
    const bool hib  = (hh != 0);
    const int  b0   = blockIdx.x * 16; // 512 blocks x 16 rows = 8192

    // ---- A operands (weights) preloaded once as bf16 A-layout fragments ----
    // A-layout: lane (m + 16*hh) holds row m, k runs [8hh,8hh+8) and [16+8hh,+8)
    v16bf A1[8];                       // W1 (128x32): 8 M-tiles, K = 32
#pragma unroll
    for (int mt = 0; mt < 8; ++mt) {
        const float* r0 = &W1[(mt * 16 + n) * D_DIM + 8 * hh];
        const float* r1 = &W1[(mt * 16 + n) * D_DIM + 16 + 8 * hh];
        Frag f;
#pragma unroll
        for (int j = 0; j < 8; ++j) {
            f.v[j]     = (__bf16)r0[j];
            f.v[8 + j] = (__bf16)r1[j];
        }
        A1[mt] = f.v;
    }
    v16bf A2[2][4];                    // W2 (32x128): 2 M-tiles x 4 K-tiles
#pragma unroll
    for (int mt = 0; mt < 2; ++mt)
#pragma unroll
        for (int kt = 0; kt < 4; ++kt) {
            const float* r0 = &W2[(mt * 16 + n) * H_DIM + kt * 32 + 8 * hh];
            const float* r1 = r0 + 16;
            Frag f;
#pragma unroll
            for (int j = 0; j < 8; ++j) {
                f.v[j]     = (__bf16)r0[j];
                f.v[8 + j] = (__bf16)r1[j];
            }
            A2[mt][kt] = f.v;
        }

    // ---- bias fragments in C-layout (lane holds m = base + 8*hh + r) ----
    v8f b1f[8], b2f[2];
#pragma unroll
    for (int mt = 0; mt < 8; ++mt)
#pragma unroll
        for (int r = 0; r < 8; ++r)
            b1f[mt][r] = b1[mt * 16 + 8 * hh + r];
#pragma unroll
    for (int mt = 0; mt < 2; ++mt)
#pragma unroll
        for (int r = 0; r < 8; ++r)
            b2f[mt][r] = b2[mt * 16 + 8 * hh + r];

    // ---- initial state y0 = x[:,0,:] in transposed C-layout ----
    // lane holds y[b0+n][d] for d = 8*hh + r (ya) and d = 16 + 8*hh + r (yb)
    const float* xrow = &x[(b0 + n) * TD];
    float*       orow = &out[(b0 + n) * TD];
    v8f ya, yb;
#pragma unroll
    for (int r = 0; r < 8; ++r) {
        ya[r] = xrow[8 * hh + r];
        yb[r] = xrow[16 + 8 * hh + r];
    }
    {   // pred[:,0,:] = y0 ; 8 contiguous floats per tile -> 2x b128 NT stores
        union { v8f v; v4f q[2]; } s0, s1;
        s0.v = ya; s1.v = yb;
        __builtin_nontemporal_store(s0.q[0], (v4f*)&orow[8 * hh]);
        __builtin_nontemporal_store(s0.q[1], (v4f*)&orow[8 * hh + 4]);
        __builtin_nontemporal_store(s1.q[0], (v4f*)&orow[16 + 8 * hh]);
        __builtin_nontemporal_store(s1.q[1], (v4f*)&orow[16 + 8 * hh + 4]);
    }

    // ---- f(u): h^T = W1 @ u^T + b1 ; out^T = W2 @ tanh(h^T) + b2 ----
    // Phased so the scheduler can cover WMMA->VALU hazard windows with
    // independent tanh/pack work from other tiles instead of v_nops.
    auto evalf = [&](v8f u0, v8f u1, v8f& o0, v8f& o1) {
        v16bf B1 = build_B(u0, u1, hib);           // u as 32x16 bf16 B operand

        // Phase 1: all 8 GEMM1 WMMAs (mutually independent, bias in C operand)
        v8f c[8];
#pragma unroll
        for (int mt = 0; mt < 8; ++mt)
            c[mt] = __builtin_amdgcn_wmma_f32_16x16x32_bf16(
                        false, A1[mt], false, B1, (short)0, b1f[mt], false, false);

        // Phase 2: tanh on all hidden tiles
#pragma unroll
        for (int mt = 0; mt < 8; ++mt)
#pragma unroll
            for (int r = 0; r < 8; ++r)
                c[mt][r] = fast_tanh(c[mt][r]);

        // Phase 3: build all 4 GEMM2 B fragments (tiles 2kt,2kt+1 -> k in [32kt,+32))
        v16bf B2[4];
#pragma unroll
        for (int kt = 0; kt < 4; ++kt)
            B2[kt] = build_B(c[2 * kt], c[2 * kt + 1], hib);

        // Phase 4: two independent K-chained accumulation streams
        v8f acc0 = b2f[0], acc1 = b2f[1];
#pragma unroll
        for (int kt = 0; kt < 4; ++kt) {
            acc0 = __builtin_amdgcn_wmma_f32_16x16x32_bf16(
                       false, A2[0][kt], false, B2[kt], (short)0, acc0, false, false);
            acc1 = __builtin_amdgcn_wmma_f32_16x16x32_bf16(
                       false, A2[1][kt], false, B2[kt], (short)0, acc1, false, false);
        }
        o0 = acc0; o1 = acc1;
    };

#pragma unroll 1
    for (int s = 0; s < T_LEN - 1; ++s) {
        const float h = tvec[s + 1] - tvec[s];

        v8f k1a, k1b, k2a, k2b, k3a, k3b, k4a, k4b, k5a, k5b, k6a, k6b, u0, u1;

        evalf(ya, yb, k1a, k1b);

        u0 = ya + k1a * (h * 0.2f);
        u1 = yb + k1b * (h * 0.2f);
        evalf(u0, u1, k2a, k2b);

        u0 = ya + (k1a * (3.0f / 40.0f) + k2a * (9.0f / 40.0f)) * h;
        u1 = yb + (k1b * (3.0f / 40.0f) + k2b * (9.0f / 40.0f)) * h;
        evalf(u0, u1, k3a, k3b);

        u0 = ya + (k1a * (44.0f / 45.0f) - k2a * (56.0f / 15.0f) + k3a * (32.0f / 9.0f)) * h;
        u1 = yb + (k1b * (44.0f / 45.0f) - k2b * (56.0f / 15.0f) + k3b * (32.0f / 9.0f)) * h;
        evalf(u0, u1, k4a, k4b);

        u0 = ya + (k1a * (19372.0f / 6561.0f) - k2a * (25360.0f / 2187.0f)
                 + k3a * (64448.0f / 6561.0f) - k4a * (212.0f / 729.0f)) * h;
        u1 = yb + (k1b * (19372.0f / 6561.0f) - k2b * (25360.0f / 2187.0f)
                 + k3b * (64448.0f / 6561.0f) - k4b * (212.0f / 729.0f)) * h;
        evalf(u0, u1, k5a, k5b);

        u0 = ya + (k1a * (9017.0f / 3168.0f) - k2a * (355.0f / 33.0f)
                 + k3a * (46732.0f / 5247.0f) + k4a * (49.0f / 176.0f)
                 - k5a * (5103.0f / 18656.0f)) * h;
        u1 = yb + (k1b * (9017.0f / 3168.0f) - k2b * (355.0f / 33.0f)
                 + k3b * (46732.0f / 5247.0f) + k4b * (49.0f / 176.0f)
                 - k5b * (5103.0f / 18656.0f)) * h;
        evalf(u0, u1, k6a, k6b);

        ya = ya + (k1a * (35.0f / 384.0f) + k3a * (500.0f / 1113.0f)
                 + k4a * (125.0f / 192.0f) - k5a * (2187.0f / 6784.0f)
                 + k6a * (11.0f / 84.0f)) * h;
        yb = yb + (k1b * (35.0f / 384.0f) + k3b * (500.0f / 1113.0f)
                 + k4b * (125.0f / 192.0f) - k5b * (2187.0f / 6784.0f)
                 + k6b * (11.0f / 84.0f)) * h;

        // pred[:, s+1, :] -- contiguous 32B per tile, non-temporal
        union { v8f v; v4f q[2]; } s0, s1;
        s0.v = ya; s1.v = yb;
        float* op = &orow[(s + 1) * 32];
        __builtin_nontemporal_store(s0.q[0], (v4f*)&op[8 * hh]);
        __builtin_nontemporal_store(s0.q[1], (v4f*)&op[8 * hh + 4]);
        __builtin_nontemporal_store(s1.q[0], (v4f*)&op[16 + 8 * hh]);
        __builtin_nontemporal_store(s1.q[1], (v4f*)&op[16 + 8 * hh + 4]);
    }
}

extern "C" void kernel_launch(void* const* d_in, const int* in_sizes, int n_in,
                              void* d_out, int out_size, void* d_ws, size_t ws_size,
                              hipStream_t stream) {
    // setup_inputs order: x, t, itv, itv_mask, W1, b1, W2, b2
    const float* x  = (const float*)d_in[0];
    const float* tv = (const float*)d_in[1];
    const float* W1 = (const float*)d_in[4];
    const float* b1 = (const float*)d_in[5];
    const float* W2 = (const float*)d_in[6];
    const float* b2 = (const float*)d_in[7];
    float* out = (float*)d_out;

    dim3 grid(B_TOT / 16);   // 512 independent wave-sized workgroups
    dim3 block(32);          // one wave32 each: no LDS, no barriers
    node_dopri5_kernel<<<grid, block, 0, stream>>>(x, tv, W1, b1, W2, b2, out);
}
